// Attention_15968688407250
// MI455X (gfx1250) — matrix operand order repaired and checked by reading, not compiled, not run on hardware
//
#include <hip/hip_runtime.h>

// ---------------------------------------------------------------------------
// GPT-2 attention block for MI455X (gfx1250), bf16 WMMA + TDM async loads.
//   B=2, S=2048, D=1024, H=16, Dh=64
//   out = [ a : B*S*D f32 ][ present : B*2*H*S*Dh f32 ]
// ---------------------------------------------------------------------------

typedef __bf16 bf16;
typedef __attribute__((ext_vector_type(8)))  __bf16 v8bf;
typedef __attribute__((ext_vector_type(16))) __bf16 v16bf;
typedef __attribute__((ext_vector_type(8)))  float  v8f;
typedef unsigned int u32x4 __attribute__((ext_vector_type(4)));
typedef int          i32x4 __attribute__((ext_vector_type(4)));
typedef int          i32x8 __attribute__((ext_vector_type(8)));

#define B_  2
#define S_  2048
#define D_  1024
#define H_  16
#define DH_ 64

// B-panel staged in LDS per block: 64 rows x 256 K-elements (bf16) = 32 KB
#define PANEL_K 256
#define PANEL_N 64

static constexpr size_t A_ELEMS = (size_t)B_ * S_ * D_;   // 4,194,304 (a region of d_out)

// ---- workspace layout (bytes) ----
static constexpr size_t WS_WQKV = 0;                       // bf16 [3072,1024]  6 MB
static constexpr size_t WS_WPRJ = 6u  << 20;               // bf16 [1024,1024]  2 MB
static constexpr size_t WS_QBF  = 8u  << 20;               // bf16 [B,H,S,Dh]   8 MB
static constexpr size_t WS_KBF  = 16u << 20;               // bf16 [B,H,S,Dh]   8 MB
static constexpr size_t WS_VT   = 24u << 20;               // bf16 [B,H,Dh,S]   8 MB
static constexpr size_t WS_AWS  = 32u << 20;               // f32  [B*S,D]     16 MB

#if defined(__has_builtin)
#if __has_builtin(__builtin_amdgcn_tensor_load_to_lds)
#define HAVE_TDM 1
#else
#define HAVE_TDM 0
#endif
#else
#define HAVE_TDM 0
#endif

// ---------------------------------------------------------------------------
// fragment helpers (16-bit WMMA VGPR layouts, CDNA5 ISA §7.12.2)
// ---------------------------------------------------------------------------
__device__ __forceinline__ v16bf cat8(v8bf a, v8bf b) {
    return __builtin_shufflevector(a, b, 0,1,2,3,4,5,6,7,8,9,10,11,12,13,14,15);
}

// A fragment: lane (L / L+16) holds row L; low half K chunks {0..7,16..23},
// high half {8..15,24..31}. rowPtr = &src[row][k0] (16B aligned).
__device__ __forceinline__ v16bf load_a_bf16(const bf16* rowPtr, int hi) {
    v8bf lo = *(const v8bf*)(rowPtr + hi * 8);
    v8bf hh = *(const v8bf*)(rowPtr + 16 + hi * 8);
    return cat8(lo, hh);
}

// B fragment: lane holds column; low half K chunks {0..7,16..23},
// high half {8..15,24..31} in the (N x K) transposed storage. rowPtr = &srcT[col][k0].
__device__ __forceinline__ v16bf load_b_bf16(const bf16* rowPtr, int hi) {
    return cat8(*(const v8bf*)(rowPtr + hi * 8), *(const v8bf*)(rowPtr + 16 + hi * 8));
}

__device__ __forceinline__ v8bf cvt8(const float* p) {
    const float4* q = (const float4*)p;
    float4 a = q[0], b = q[1];
    v8bf r;
    r[0] = (bf16)a.x; r[1] = (bf16)a.y; r[2] = (bf16)a.z; r[3] = (bf16)a.w;
    r[4] = (bf16)b.x; r[5] = (bf16)b.y; r[6] = (bf16)b.z; r[7] = (bf16)b.w;
    return r;
}

// A fragment sourced from f32 row-major, converted on the fly.
__device__ __forceinline__ v16bf load_a_f32(const float* rowPtr, int hi) {
    return cat8(cvt8(rowPtr + hi * 8), cvt8(rowPtr + 16 + hi * 8));
}

__device__ __forceinline__ v8f wmma_bf16(v16bf a, v16bf b, v8f c) {
    return __builtin_amdgcn_wmma_f32_16x16x32_bf16(
        /*neg_a=*/false, a, /*neg_b=*/false, b,
        /*c_mod=*/(short)0, c, /*reuse_a=*/false, /*reuse_b=*/false);
}

__device__ __forceinline__ v8f vzero8() {
    v8f z = {0.f, 0.f, 0.f, 0.f, 0.f, 0.f, 0.f, 0.f};
    return z;
}

// ---------------------------------------------------------------------------
// TDM: DMA a 2D bf16 tile (PANEL_N rows x tile_k elems, row stride
// row_stride_elems) from global into LDS at byte offset lds_off.
// Descriptor per CDNA5 ISA (D# groups 0/1; groups 2/3
// zero => tile_dim3/4 unused).
// ---------------------------------------------------------------------------
#if HAVE_TDM
__device__ __forceinline__ void tdm_load_panel(unsigned lds_off, const bf16* gptr,
                                               unsigned tile_k, unsigned tile_rows,
                                               unsigned row_stride_elems) {
    unsigned long long ga = (unsigned long long)(uintptr_t)gptr;
    u32x4 g0;
    g0[0] = 1u;                                       // count=1, user mode
    g0[1] = lds_off;                                  // lds_addr (bytes)
    g0[2] = (unsigned)(ga & 0xFFFFFFFFu);             // global_addr[31:0]
    g0[3] = (unsigned)((ga >> 32) & 0x1FFFFFFu)       // global_addr[56:32]
          | (2u << 30);                               // type = 2 ("image")
    i32x8 g1;
    g1[0] = (int)(1u << 16);                          // data_size = 1 (2 bytes)
    g1[1] = (int)(tile_k << 16);                      // tensor_dim0[15:0]
    g1[2] = (int)(tile_rows << 16);                   // tensor_dim1[15:0]
    g1[3] = (int)(tile_k << 16);                      // tile_dim0
    g1[4] = (int)tile_rows;                           // tile_dim1 (tile_dim2=0)
    g1[5] = (int)row_stride_elems;                    // tensor_dim0_stride[31:0]
    g1[6] = 0;
    g1[7] = 0;
    i32x4 z4 = {0, 0, 0, 0};
#if __has_include(<hip/amd_detail/amd_gfx1250_TDM.h>)
    i32x8 z8 = {0, 0, 0, 0, 0, 0, 0, 0};
    __builtin_amdgcn_tensor_load_to_lds(g0, g1, z4, z4, z8, 0);
#else
    __builtin_amdgcn_tensor_load_to_lds(g0, g1, z4, z4, 0);
#endif
}
#endif

// Stage one K-phase of the weight panel into LDS. TDM path: single DMA issued
// by wave 0 + s_wait_tensorcnt. Fallback: cooperative vector copy.
__device__ __forceinline__ void stage_panel(bf16* ldsB, const bf16* gsrc, int tid) {
#if HAVE_TDM
    if (tid < 32) {
        tdm_load_panel((unsigned)(uintptr_t)ldsB, gsrc, PANEL_K, PANEL_N, D_);
        __builtin_amdgcn_s_wait_tensorcnt(0);
    }
#else
    for (int c = tid; c < (PANEL_N * PANEL_K) / 8; c += 256) {
        int row = c >> 5;            // 32 8-elem chunks per 256-elem row
        int col = (c & 31) * 8;
        *(v8bf*)&ldsB[row * PANEL_K + col] =
            *(const v8bf*)&gsrc[(size_t)row * D_ + col];
    }
#endif
}

// ---------------------------------------------------------------------------
// Kernel 0: convert+transpose fp32 weight [K,N] -> bf16 [N,K]
// ---------------------------------------------------------------------------
__global__ void cvt_transpose_kernel(const float* __restrict__ W,
                                     bf16* __restrict__ Wt, int K, int N) {
    size_t idx = (size_t)blockIdx.x * 256 + threadIdx.x;
    if (idx >= (size_t)N * K) return;
    int n = (int)(idx / (size_t)K);
    int k = (int)(idx % (size_t)K);
    Wt[idx] = (bf16)W[(size_t)k * N + n];
}

// ---------------------------------------------------------------------------
// Kernel 1: QKV GEMM  [4096,1024] x [1024,3072] + bias.
// Block = one 64-wide N panel x 8 m-tiles (one per wave). Weight panel is
// TDM-DMA'd into LDS per 256-wide K-phase and shared by all 8 waves.
//   q -> bf16 [B,H,S,Dh] scaled by 1/sqrt(Dh)
//   k -> present f32 [b,0,h,s,dh] and bf16 [B,H,S,Dh]
//   v -> present f32 [b,1,h,s,dh] and bf16 transposed [B,H,Dh,S]
// ---------------------------------------------------------------------------
__global__ __launch_bounds__(256) void qkv_gemm_kernel(
    const float* __restrict__ x, const bf16* __restrict__ wt,
    const float* __restrict__ bias, float* __restrict__ out,
    bf16* __restrict__ q_bf, bf16* __restrict__ k_bf, bf16* __restrict__ v_t) {

    __shared__ bf16 ldsB[PANEL_N * PANEL_K];   // 32 KB weight panel

    const int tid  = threadIdx.x;
    const int lane = tid & 31;
    const int wave = tid >> 5;
    const int hi = lane >> 4;
    const int ln = lane & 15;

    int nt = blockIdx.x >> 5;                 // 0..47  (N = 3072)
    int mg = blockIdx.x & 31;                 // 0..31  (M groups of 8 tiles)
    int n0 = nt * 64;
    int m0 = (mg * 8 + wave) * 16;

    v8f acc[4] = {vzero8(), vzero8(), vzero8(), vzero8()};

    const float* aRow = x + (size_t)(m0 + ln) * D_;

    for (int p = 0; p < D_ / PANEL_K; ++p) {
        const int kb = p * PANEL_K;
        __syncthreads();                       // previous panel fully consumed
        stage_panel(ldsB, wt + (size_t)n0 * D_ + kb, tid);
        __syncthreads();                       // panel visible to all waves

        for (int k0 = 0; k0 < PANEL_K; k0 += 32) {
            __builtin_prefetch(aRow + kb + k0 + 64, 0, 0);
            v16bf a = load_a_f32(aRow + kb + k0, hi);
#pragma unroll
            for (int t = 0; t < 4; ++t) {
                v16bf b = load_b_bf16(&ldsB[(t * 16 + ln) * PANEL_K + k0], hi);
                acc[t] = wmma_bf16(a, b, acc[t]);
            }
        }
    }

#pragma unroll
    for (int t = 0; t < 4; ++t) {
        int f = n0 + t * 16 + ln;
        float bv = bias[f];
#pragma unroll
        for (int r = 0; r < 8; ++r) {
            int m = m0 + r + 8 * hi;
            int s = m & (S_ - 1);
            int bb = m >> 11;
            float val = acc[t][r] + bv;
            if (f < D_) {                       // q
                int h = f >> 6, dh = f & 63;
                q_bf[(((size_t)bb * H_ + h) * S_ + s) * DH_ + dh] =
                    (bf16)(val * 0.125f);       // fold 1/sqrt(64)
            } else if (f < 2 * D_) {            // k
                int g = f - D_;
                int h = g >> 6, dh = g & 63;
                size_t pidx = ((((size_t)bb * 2 + 0) * H_ + h) * S_ + s) * DH_ + dh;
                out[A_ELEMS + pidx] = val;
                k_bf[(((size_t)bb * H_ + h) * S_ + s) * DH_ + dh] = (bf16)val;
            } else {                            // v
                int g = f - 2 * D_;
                int h = g >> 6, dh = g & 63;
                size_t pidx = ((((size_t)bb * 2 + 1) * H_ + h) * S_ + s) * DH_ + dh;
                out[A_ELEMS + pidx] = val;
                v_t[(((size_t)bb * H_ + h) * DH_ + dh) * S_ + s] = (bf16)val;
            }
        }
    }
}

// ---------------------------------------------------------------------------
// Kernel 2: causal flash attention, bf16 WMMA, online softmax.
// grid (128 q-tiles, 4), block = 8 waves; all waves in a block share the
// q-tile (uniform trip count -> __syncthreads is safe), wave i owns head
// bh = blockIdx.y*8+i. Per 32-key block: 4 WMMA (QK^T) + 4 WMMA (PV).
// ---------------------------------------------------------------------------
__global__ __launch_bounds__(256) void attn_kernel(
    const bf16* __restrict__ q_bf, const bf16* __restrict__ k_bf,
    const bf16* __restrict__ v_t, float* __restrict__ a_ws) {

    __shared__ bf16 ldsP[8][16 * 32];   // per-wave P tile (16 x 32), 8 KB total

    const int lane = threadIdx.x & 31;
    const int wave = threadIdx.x >> 5;
    const int hi = lane >> 4;
    const int ln = lane & 15;

    int qt = blockIdx.x;                    // 0..127
    int bh = blockIdx.y * 8 + wave;         // 0..31
    int s0 = qt * 16;

    const bf16* Q = q_bf + (size_t)bh * S_ * DH_;
    const bf16* K = k_bf + (size_t)bh * S_ * DH_;
    const bf16* V = v_t + (size_t)bh * DH_ * S_;

    float mrow[8], lrow[8];
    v8f acc[4] = {vzero8(), vzero8(), vzero8(), vzero8()};
#pragma unroll
    for (int r = 0; r < 8; ++r) { mrow[r] = -3.0e38f; lrow[r] = 0.f; }

    // Q fragments (16 x 64) loaded once; scale already folded in.
    v16bf qa0 = load_a_bf16(Q + (size_t)(s0 + ln) * DH_ + 0, hi);
    v16bf qa1 = load_a_bf16(Q + (size_t)(s0 + ln) * DH_ + 32, hi);

    const int kend = s0 + 16;               // causal: keys < kend
    for (int j = 0; j < kend; j += 32) {
        // ---- scores: S[16 x 32] = Q (16x64) @ K^T -------------------------
        v8f sc0 = vzero8(), sc1 = vzero8();
        {
            v16bf bk;
            bk = load_b_bf16(K + (size_t)(j + ln) * DH_ + 0, hi);
            sc0 = wmma_bf16(qa0, bk, sc0);
            bk = load_b_bf16(K + (size_t)(j + ln) * DH_ + 32, hi);
            sc0 = wmma_bf16(qa1, bk, sc0);
            bk = load_b_bf16(K + (size_t)(j + 16 + ln) * DH_ + 0, hi);
            sc1 = wmma_bf16(qa0, bk, sc1);
            bk = load_b_bf16(K + (size_t)(j + 16 + ln) * DH_ + 32, hi);
            sc1 = wmma_bf16(qa1, bk, sc1);
        }

        // ---- causal mask (exactly -1e10 like the reference) ---------------
        if (j + 31 > s0) {
#pragma unroll
            for (int r = 0; r < 8; ++r) {
                int qrow = s0 + r + 8 * hi;
                if (j + ln > qrow)      sc0[r] = -1.0e10f;
                if (j + 16 + ln > qrow) sc1[r] = -1.0e10f;
            }
        }

        // ---- online softmax; rows live at (VGPR r, half hi) in C-layout ---
#pragma unroll
        for (int r = 0; r < 8; ++r) {
            float mx = fmaxf(sc0[r], sc1[r]);
#pragma unroll
            for (int d = 1; d < 16; d <<= 1) mx = fmaxf(mx, __shfl_xor(mx, d));
            float mn = fmaxf(mrow[r], mx);
            float alpha = __expf(mrow[r] - mn);
            float p0 = __expf(sc0[r] - mn);
            float p1 = __expf(sc1[r] - mn);
            float sum = p0 + p1;
#pragma unroll
            for (int d = 1; d < 16; d <<= 1) sum += __shfl_xor(sum, d);
            lrow[r] = lrow[r] * alpha + sum;
            mrow[r] = mn;
#pragma unroll
            for (int t = 0; t < 4; ++t) acc[t][r] *= alpha;
            int row = r + 8 * hi;
            ldsP[wave][row * 32 + ln]      = (bf16)p0;
            ldsP[wave][row * 32 + 16 + ln] = (bf16)p1;
        }
        __syncthreads();

        // ---- C-layout -> A-layout transpose via LDS, then PV --------------
        v8bf plo = *(const v8bf*)&ldsP[wave][ln * 32 + hi * 8];
        v8bf phh = *(const v8bf*)&ldsP[wave][ln * 32 + 16 + hi * 8];
        v16bf pa = cat8(plo, phh);
        __syncthreads();

#pragma unroll
        for (int t = 0; t < 4; ++t) {
            v16bf bv = load_b_bf16(V + (size_t)(t * 16 + ln) * S_ + j, hi);
            acc[t] = wmma_bf16(pa, bv, acc[t]);
        }
    }

    // ---- normalize and write merged-heads activation [B,S,D] --------------
    int b = bh >> 4, h = bh & 15;
#pragma unroll
    for (int r = 0; r < 8; ++r) {
        float inv = 1.0f / lrow[r];
        int s = s0 + r + 8 * hi;
        float* dst = a_ws + ((size_t)(b * S_ + s)) * D_ + h * DH_;
#pragma unroll
        for (int t = 0; t < 4; ++t) dst[t * 16 + ln] = acc[t][r] * inv;
    }
}

// ---------------------------------------------------------------------------
// Kernel 3: output projection [4096,1024] x [1024,1024] + bias -> d_out.
// Same TDM-staged panel scheme as the QKV GEMM.
// ---------------------------------------------------------------------------
__global__ __launch_bounds__(256) void proj_gemm_kernel(
    const float* __restrict__ a_ws, const bf16* __restrict__ wt,
    const float* __restrict__ bias, float* __restrict__ out) {

    __shared__ bf16 ldsB[PANEL_N * PANEL_K];   // 32 KB weight panel

    const int tid  = threadIdx.x;
    const int lane = tid & 31;
    const int wave = tid >> 5;
    const int hi = lane >> 4;
    const int ln = lane & 15;

    int nt = blockIdx.x >> 5;                 // 0..15  (N = 1024)
    int mg = blockIdx.x & 31;                 // 0..31
    int n0 = nt * 64;
    int m0 = (mg * 8 + wave) * 16;

    v8f acc[4] = {vzero8(), vzero8(), vzero8(), vzero8()};

    const float* aRow = a_ws + (size_t)(m0 + ln) * D_;

    for (int p = 0; p < D_ / PANEL_K; ++p) {
        const int kb = p * PANEL_K;
        __syncthreads();
        stage_panel(ldsB, wt + (size_t)n0 * D_ + kb, tid);
        __syncthreads();

        for (int k0 = 0; k0 < PANEL_K; k0 += 32) {
            __builtin_prefetch(aRow + kb + k0 + 64, 0, 0);
            v16bf a = load_a_f32(aRow + kb + k0, hi);
#pragma unroll
            for (int t = 0; t < 4; ++t) {
                v16bf b = load_b_bf16(&ldsB[(t * 16 + ln) * PANEL_K + k0], hi);
                acc[t] = wmma_bf16(a, b, acc[t]);
            }
        }
    }

#pragma unroll
    for (int t = 0; t < 4; ++t) {
        int f = n0 + t * 16 + ln;
        float bv = bias[f];
#pragma unroll
        for (int r = 0; r < 8; ++r) {
            int m = m0 + r + 8 * hi;
            out[(size_t)m * D_ + f] = acc[t][r] + bv;
        }
    }
}

// ---------------------------------------------------------------------------
extern "C" void kernel_launch(void* const* d_in, const int* in_sizes, int n_in,
                              void* d_out, int out_size, void* d_ws, size_t ws_size,
                              hipStream_t stream) {
    (void)in_sizes; (void)n_in; (void)out_size; (void)ws_size;

    const float* x        = (const float*)d_in[0];
    const float* c_attn_w = (const float*)d_in[1];
    const float* c_attn_b = (const float*)d_in[2];
    const float* c_proj_w = (const float*)d_in[3];
    const float* c_proj_b = (const float*)d_in[4];
    float* out = (float*)d_out;

    char* ws = (char*)d_ws;
    bf16*  wqkv_t = (bf16*)(ws + WS_WQKV);
    bf16*  wp_t   = (bf16*)(ws + WS_WPRJ);
    bf16*  q_bf   = (bf16*)(ws + WS_QBF);
    bf16*  k_bf   = (bf16*)(ws + WS_KBF);
    bf16*  v_t    = (bf16*)(ws + WS_VT);
    float* a_ws   = (float*)(ws + WS_AWS);

    // weights: fp32 [K,N] -> bf16 [N,K]
    cvt_transpose_kernel<<<(3 * D_ * D_ + 255) / 256, 256, 0, stream>>>(
        c_attn_w, wqkv_t, D_, 3 * D_);
    cvt_transpose_kernel<<<(D_ * D_ + 255) / 256, 256, 0, stream>>>(
        c_proj_w, wp_t, D_, D_);

    // QKV projection (+ present outputs, bf16 staging)
    qkv_gemm_kernel<<<1536, 256, 0, stream>>>(x, wqkv_t, c_attn_b, out,
                                              q_bf, k_bf, v_t);

    // causal flash attention
    attn_kernel<<<dim3(128, 4), 256, 0, stream>>>(q_bf, k_bf, v_t, a_ws);

    // output projection
    proj_gemm_kernel<<<512, 256, 0, stream>>>(a_ws, wp_t, c_proj_b, out);
}
